// CapsNet_11149735100983
// MI455X (gfx1250) — matrix operand order repaired and checked by reading, not compiled, not run on hardware
//
#include <hip/hip_runtime.h>
#include <hip/hip_fp16.h>
#include <hip/hip_bf16.h>

typedef __attribute__((ext_vector_type(16))) _Float16 v16h;
typedef __attribute__((ext_vector_type(8)))  _Float16 v8h;
typedef __attribute__((ext_vector_type(8)))  float    v8f;

#define TM 64
#define TN 64
#define TK 32

union Pk4 { _Float16 f[4]; unsigned long long u; };

// =====================================================================
// WMMA GEMM core: C[M,N] = A[M,K] @ B[K,N] (+bias/+gelu/+residual)
// fp32 in memory, f16 fragment-swizzled LDS staging, f32 WMMA accum.
// Block = 128 threads = 4 waves; wave owns a 16-row strip (4 wmma/step).
// LDS layout is fragment-ready: [frag][kchunk][lane][8 halves], so a
// fragment load is 2x ds_load_b128 per matrix, no per-element packing.
// EPI: 0=none, 1=bias, 2=bias+exact gelu, 3=bias+residual
// Requires K % 4 == 0 and all leading dims % 4 == 0 (true for all uses).
// =====================================================================
template<bool TRANSB, int EPI>
__device__ __forceinline__ void gemm_tile(
    const float* __restrict__ A, int lda,
    const float* __restrict__ Bm, int ldb,
    float* __restrict__ C, int ldc,
    const float* __restrict__ bias,
    const float* __restrict__ resid, int ldr,
    int M, int N, int K, int tm, int tn)
{
  // [wave-strip][k-chunk][lane][8 halves] ; 4 KB each
  __shared__ __align__(16) _Float16 Asw[4][2][32][8];
  __shared__ __align__(16) _Float16 Bsw[4][2][32][8];
  const int tid  = threadIdx.x;      // 0..127
  const int lane = tid & 31;
  const int wave = tid >> 5;

  v8f acc[4];
  #pragma unroll
  for (int i = 0; i < 4; ++i)
    #pragma unroll
    for (int j = 0; j < 8; ++j) acc[i][j] = 0.0f;

  const int row0 = tm * TM;
  const int col0 = tn * TN;

  for (int k0 = 0; k0 < K; k0 += TK) {
    // ---- stage A tile: 64 rows x 8 k-quads; float4 load -> ds_store_b64 ----
    #pragma unroll
    for (int q = 0; q < 4; ++q) {
      int idx = q * 128 + tid;
      int r = idx >> 3, c = (idx & 7) * 4;
      int gr = row0 + r, gc = k0 + c;
      float4 v = make_float4(0.f, 0.f, 0.f, 0.f);
      if (gr < M && gc < K) v = *(const float4*)&A[(long)gr * lda + gc];
      Pk4 p;
      p.f[0] = (_Float16)v.x; p.f[1] = (_Float16)v.y;
      p.f[2] = (_Float16)v.z; p.f[3] = (_Float16)v.w;
      // fragment address: k = (h&7) + ((h>>3)<<4) + (grp<<3)
      int grp = (c >> 3) & 1;
      *(unsigned long long*)&Asw[r >> 4][(c >> 4) & 1][(r & 15) + 16 * grp][c & 7] = p.u;
    }
    // ---- stage B tile: 8 k-quads x 64 cols ----
    #pragma unroll
    for (int q = 0; q < 4; ++q) {
      int idx = q * 128 + tid;
      int n = idx & 63, kk = (idx >> 6) * 4;
      int gk = k0 + kk, gn = col0 + n;
      float4 v = make_float4(0.f, 0.f, 0.f, 0.f);
      if (gk < K && gn < N) {
        if (TRANSB) {
          v = *(const float4*)&Bm[(long)gn * ldb + gk];
        } else {
          v.x = Bm[(long)gk * ldb + gn];
          v.y = Bm[(long)(gk + 1) * ldb + gn];
          v.z = Bm[(long)(gk + 2) * ldb + gn];
          v.w = Bm[(long)(gk + 3) * ldb + gn];
        }
      }
      Pk4 p;
      p.f[0] = (_Float16)v.x; p.f[1] = (_Float16)v.y;
      p.f[2] = (_Float16)v.z; p.f[3] = (_Float16)v.w;
      // fragment address: k = (lane>>4)*8 + chunk*16 + slot
      *(unsigned long long*)&Bsw[n >> 4][(kk >> 4) & 1][(n & 15) + 16 * ((kk >> 3) & 1)][kk & 7] = p.u;
    }
    // prefetch next K-tile of A (speculative, CDNA5 global_prefetch)
    if (k0 + TK < K) {
      int pr = row0 + (tid >> 1); if (pr >= M) pr = M - 1;
      __builtin_prefetch(&A[(long)pr * lda + k0 + TK], 0, 0);
    }
    __syncthreads();

    // ---- fragment loads: 2x ds_load_b128 each, zero repacking ----
    v8h a0 = *(const v8h*)&Asw[wave][0][lane][0];
    v8h a1 = *(const v8h*)&Asw[wave][1][lane][0];
    v16h afrag = __builtin_shufflevector(a0, a1, 0, 1, 2, 3, 4, 5, 6, 7,
                                         8, 9, 10, 11, 12, 13, 14, 15);
    #pragma unroll
    for (int nt = 0; nt < 4; ++nt) {
      v8h b0 = *(const v8h*)&Bsw[nt][0][lane][0];
      v8h b1 = *(const v8h*)&Bsw[nt][1][lane][0];
      v16h bfrag = __builtin_shufflevector(b0, b1, 0, 1, 2, 3, 4, 5, 6, 7,
                                           8, 9, 10, 11, 12, 13, 14, 15);
      acc[nt] = __builtin_amdgcn_wmma_f32_16x16x32_f16(
          false, afrag, false, bfrag, (short)0, acc[nt], false, false);
    }
    __syncthreads();
  }

  // ---- epilogue: C/D layout = 8 VGPRs, lanes 16-31 hold M+8 ----
  const int mbase = wave * 16 + (lane >> 4) * 8;
  const int ncol  = lane & 15;
  #pragma unroll
  for (int nt = 0; nt < 4; ++nt) {
    int gn = col0 + nt * 16 + ncol;
    if (gn >= N) continue;
    #pragma unroll
    for (int r = 0; r < 8; ++r) {
      int gm = row0 + mbase + r;
      if (gm >= M) continue;
      float v = acc[nt][r];
      if (EPI >= 1) v += bias[gn];
      if (EPI == 2) v = 0.5f * v * (1.0f + erff(v * 0.7071067811865475f)); // exact gelu
      if (EPI == 3) v += resid[(long)gm * ldr + gn];
      C[(long)gm * ldc + gn] = v;
    }
  }
}

template<bool TRANSB, int EPI>
__global__ void gemm_kernel(const float* A, int lda, const float* B, int ldb,
                            float* C, int ldc, const float* bias,
                            const float* resid, int ldr, int M, int N, int K) {
  gemm_tile<TRANSB, EPI>(A, lda, B, ldb, C, ldc, bias, resid, ldr,
                         M, N, K, blockIdx.x, blockIdx.y);
}

// Batched attention GEMMs: grid.z = b*12+h
__global__ void attn_scores_kernel(const float* __restrict__ qkv, float* __restrict__ S) {
  int bh = blockIdx.z, b = bh / 12, h = bh % 12;
  const float* Q  = qkv + (long)b * 180 * 2304 + h * 64;
  const float* Km = Q + 768;
  float* Sp = S + (long)bh * 180 * 180;
  gemm_tile<true, 0>(Q, 2304, Km, 2304, Sp, 180, nullptr, nullptr, 0,
                     180, 180, 64, blockIdx.x, blockIdx.y);
}

__global__ void attn_o_kernel(const float* __restrict__ S, const float* __restrict__ qkv,
                              float* __restrict__ O) {
  int bh = blockIdx.z, b = bh / 12, h = bh % 12;
  const float* P = S + (long)bh * 180 * 180;
  const float* V = qkv + (long)b * 180 * 2304 + 1536 + h * 64;
  float* Op = O + (long)b * 180 * 768 + h * 64;
  gemm_tile<false, 0>(P, 180, V, 2304, Op, 768, nullptr, nullptr, 0,
                      180, 64, 180, blockIdx.x, blockIdx.y);
}

// =====================================================================
// LayerNorm: one wave32 per 64-wide row (biased variance, eps=1e-5)
// =====================================================================
__global__ void layernorm_kernel(const float* __restrict__ X, float* __restrict__ Y,
                                 const float* __restrict__ g, const float* __restrict__ b,
                                 int nrows) {
  int row = blockIdx.x * 8 + (threadIdx.x >> 5);
  if (row >= nrows) return;
  int lane = threadIdx.x & 31;
  const float* xr = X + (long)row * 64;
  float x0 = xr[lane], x1 = xr[lane + 32];
  float s = x0 + x1, ss = x0 * x0 + x1 * x1;
  for (int o = 16; o > 0; o >>= 1) {
    s  += __shfl_xor(s, o, 32);
    ss += __shfl_xor(ss, o, 32);
  }
  float mean = s * (1.0f / 64.0f);
  float var  = ss * (1.0f / 64.0f) - mean * mean;
  float inv  = rsqrtf(var + 1e-5f);
  float* yr = Y + (long)row * 64;
  yr[lane]      = (x0 - mean) * inv * g[lane]      + b[lane];
  yr[lane + 32] = (x1 - mean) * inv * g[lane + 32] + b[lane + 32];
}

// softmax over rows of 180 with 1/sqrt(64) scale; wave per row
__global__ void softmax_kernel(float* __restrict__ S, long rows) {
  long row = (long)blockIdx.x * 8 + (threadIdx.x >> 5);
  if (row >= rows) return;
  int lane = threadIdx.x & 31;
  float* r = S + row * 180;
  float vals[6], mx = -1e30f;
  #pragma unroll
  for (int t = 0; t < 6; ++t) {
    int j = lane + t * 32;
    vals[t] = (j < 180) ? r[j] * 0.125f : -1e30f;
    mx = fmaxf(mx, vals[t]);
  }
  for (int o = 16; o > 0; o >>= 1) mx = fmaxf(mx, __shfl_xor(mx, o, 32));
  float sum = 0.0f;
  #pragma unroll
  for (int t = 0; t < 6; ++t) { vals[t] = __expf(vals[t] - mx); sum += vals[t]; }
  for (int o = 16; o > 0; o >>= 1) sum += __shfl_xor(sum, o, 32);
  float inv = 1.0f / sum;
  #pragma unroll
  for (int t = 0; t < 6; ++t) { int j = lane + t * 32; if (j < 180) r[j] = vals[t] * inv; }
}

// =====================================================================
// Direct 3x3 conv (NCHW / OIHW), optional bias & relu
// =====================================================================
__global__ void conv3x3_kernel(const float* __restrict__ in, const float* __restrict__ w,
                               const float* __restrict__ bias, float* __restrict__ out,
                               int Bn, int Cin, int Cout, int H, int W, int Ho, int Wo,
                               int stride, int pad, int dorelu) {
  long idx = (long)blockIdx.x * blockDim.x + threadIdx.x;
  long total = (long)Bn * Cout * Ho * Wo;
  if (idx >= total) return;
  int x = idx % Wo; long t = idx / Wo;
  int y = t % Ho; t /= Ho;
  int co = t % Cout; int b = t / Cout;
  float acc = bias ? bias[co] : 0.0f;
  const float* wp = w + (long)co * Cin * 9;
  const float* ip = in + (long)b * Cin * H * W;
  for (int ci = 0; ci < Cin; ++ci) {
    const float* ipc = ip + (long)ci * H * W;
    const float* wpc = wp + ci * 9;
    #pragma unroll
    for (int ky = 0; ky < 3; ++ky) {
      int iy = y * stride + ky - pad;
      if (iy < 0 || iy >= H) continue;
      #pragma unroll
      for (int kx = 0; kx < 3; ++kx) {
        int ix = x * stride + kx - pad;
        if (ix < 0 || ix >= W) continue;
        acc += ipc[iy * W + ix] * wpc[ky * 3 + kx];
      }
    }
  }
  if (dorelu) acc = fmaxf(acc, 0.0f);
  out[idx] = acc;
}

// maxpool 3x3 stride2 pad1
__global__ void maxpool_kernel(const float* __restrict__ in, float* __restrict__ out,
                               long BC, int H, int W, int Ho, int Wo) {
  long idx = (long)blockIdx.x * blockDim.x + threadIdx.x;
  long total = BC * Ho * Wo;
  if (idx >= total) return;
  int x = idx % Wo; long t = idx / Wo;
  int y = t % Ho; long bc = t / Ho;
  const float* ip = in + bc * H * W;
  float m = -3.4e38f;
  #pragma unroll
  for (int ky = 0; ky < 3; ++ky) {
    int iy = y * 2 + ky - 1;
    if (iy < 0 || iy >= H) continue;
    #pragma unroll
    for (int kx = 0; kx < 3; ++kx) {
      int ix = x * 2 + kx - 1;
      if (ix < 0 || ix >= W) continue;
      m = fmaxf(m, ip[iy * W + ix]);
    }
  }
  out[idx] = m;
}

// pooled [256,64,10,18] -> tokens X[(b*180+n),64] + pos_emb
__global__ void tokens_kernel(const float* __restrict__ pool2, const float* __restrict__ pos,
                              float* __restrict__ X) {
  long idx = (long)blockIdx.x * blockDim.x + threadIdx.x;
  if (idx >= (long)256 * 180 * 64) return;
  int d = idx & 63; long t = idx >> 6;
  int n = t % 180; int b = t / 180;
  X[idx] = pool2[((long)b * 64 + d) * 180 + n] + pos[n * 64 + d];
}

// X[(b*180+n),64] -> G[b,d,r,c] (n = r*18+c)
__global__ void g_kernel(const float* __restrict__ X, float* __restrict__ G) {
  long idx = (long)blockIdx.x * blockDim.x + threadIdx.x;
  if (idx >= (long)256 * 64 * 180) return;
  int n = idx % 180; long t = idx / 180;
  int d = t % 64; int b = t / 64;
  G[idx] = X[((long)b * 180 + n) * 64 + d];
}

// PrimaryCaps reshape + squash: pc[256,256,4,8] -> caps[256,1024,8]
__global__ void caps_kernel(const float* __restrict__ pc, float* __restrict__ caps) {
  long idx = (long)blockIdx.x * blockDim.x + threadIdx.x;
  if (idx >= (long)256 * 1024) return;
  int i = idx % 1024; int b = idx / 1024;
  int g32 = i >> 5, y = (i >> 3) & 3, x = i & 7;
  float v[8]; float l2 = 0.0f;
  #pragma unroll
  for (int e = 0; e < 8; ++e) {
    v[e] = pc[(((long)b * 256 + g32 * 8 + e) * 4 + y) * 8 + x];
    l2 += v[e] * v[e];
  }
  float sc = sqrtf(l2) / (1.0f + l2);
  #pragma unroll
  for (int e = 0; e < 8; ++e) caps[idx * 8 + e] = v[e] * sc;
}

// u[b,i,o] = sum_e caps[b,i,e] * caps_w[i,e,o]
__global__ void u_kernel(const float* __restrict__ caps, const float* __restrict__ cw,
                         float* __restrict__ u) {
  long idx = (long)blockIdx.x * blockDim.x + threadIdx.x;
  if (idx >= (long)256 * 1024 * 48) return;
  int o = idx % 48; long t = idx / 48;
  int i = t % 1024; int b = t / 1024;
  float a = 0.0f;
  #pragma unroll
  for (int e = 0; e < 8; ++e)
    a += caps[((long)b * 1024 + i) * 8 + e] * cw[((long)i * 8 + e) * 48 + o];
  u[idx] = a;
}

// ----- routing -----
__global__ void route_init(const float* __restrict__ br, float* __restrict__ bb) {
  long idx = (long)blockIdx.x * blockDim.x + threadIdx.x;
  if (idx >= (long)256 * 1024 * 3) return;
  bb[idx] = br[idx % 3072];
}
__global__ void route_c(const float* __restrict__ bb, float* __restrict__ c) {
  long idx = (long)blockIdx.x * blockDim.x + threadIdx.x;
  if (idx >= (long)256 * 1024) return;
  float b0 = bb[idx * 3], b1 = bb[idx * 3 + 1], b2 = bb[idx * 3 + 2];
  float m = fmaxf(b0, fmaxf(b1, b2));
  float e0 = __expf(b0 - m), e1 = __expf(b1 - m), e2 = __expf(b2 - m);
  float inv = 1.0f / (e0 + e1 + e2);
  c[idx * 3] = e0 * inv; c[idx * 3 + 1] = e1 * inv; c[idx * 3 + 2] = e2 * inv;
}
__global__ void route_s(const float* __restrict__ c, const float* __restrict__ u,
                        float* __restrict__ s) {
  int idx = blockIdx.x * blockDim.x + threadIdx.x;
  if (idx >= 256 * 48) return;
  int d = idx & 15; int j = (idx >> 4) % 3; int b = idx / 48;
  const float* cb = c + (long)b * 1024 * 3;
  const float* ub = u + (long)b * 1024 * 48;
  float a = 0.0f;
  for (int i = 0; i < 1024; ++i) a += cb[i * 3 + j] * ub[i * 48 + j * 16 + d];
  s[idx] = a;
}
__global__ void route_squash(const float* __restrict__ s, float* __restrict__ v) {
  int idx = blockIdx.x * blockDim.x + threadIdx.x;
  if (idx >= 768) return;
  const float* sp = s + idx * 16;
  float l2 = 0.0f;
  #pragma unroll
  for (int d = 0; d < 16; ++d) l2 += sp[d] * sp[d];
  float sc = sqrtf(l2) / (1.0f + l2);
  #pragma unroll
  for (int d = 0; d < 16; ++d) v[idx * 16 + d] = sp[d] * sc;
}
__global__ void route_bb(const float* __restrict__ u, const float* __restrict__ v,
                         float* __restrict__ bb) {
  long idx = (long)blockIdx.x * blockDim.x + threadIdx.x;
  if (idx >= (long)256 * 1024) return;
  int b = idx / 1024;
  const float* up = u + idx * 48;
  const float* vp = v + (long)b * 48;
  #pragma unroll
  for (int j = 0; j < 3; ++j) {
    float a = 0.0f;
    #pragma unroll
    for (int d = 0; d < 16; ++d) a += up[j * 16 + d] * vp[j * 16 + d];
    bb[idx * 3 + j] += a;
  }
}
__global__ void vnorm_kernel(const float* __restrict__ v, float* __restrict__ out) {
  int idx = blockIdx.x * blockDim.x + threadIdx.x;
  if (idx >= 768) return;
  float l2 = 0.0f;
  #pragma unroll
  for (int d = 0; d < 16; ++d) { float t = v[idx * 16 + d]; l2 += t * t; }
  out[idx] = sqrtf(l2);
}

// =====================================================================
// Host launcher
// =====================================================================
static inline long cdiv(long a, long b) { return (a + b - 1) / b; }

extern "C" void kernel_launch(void* const* d_in, const int* in_sizes, int n_in,
                              void* d_out, int out_size, void* d_ws, size_t ws_size,
                              hipStream_t stream) {
  const float* x       = (const float*)d_in[0];
  const float* conv1_w = (const float*)d_in[1];
  const float* conv2_w = (const float*)d_in[2];
  const float* pos_emb = (const float*)d_in[3];
  const float* ln1_g   = (const float*)d_in[4];
  const float* ln1_b   = (const float*)d_in[5];
  const float* qkv_w   = (const float*)d_in[6];
  const float* out_w   = (const float*)d_in[7];
  const float* out_b   = (const float*)d_in[8];
  const float* ln2_g   = (const float*)d_in[9];
  const float* ln2_b   = (const float*)d_in[10];
  const float* ff1_w   = (const float*)d_in[11];
  const float* ff1_b   = (const float*)d_in[12];
  const float* ff2_w   = (const float*)d_in[13];
  const float* ff2_b   = (const float*)d_in[14];
  const float* pcaps_w = (const float*)d_in[15];
  const float* pcaps_b = (const float*)d_in[16];
  const float* caps_w  = (const float*)d_in[17];
  const float* b_route = (const float*)d_in[18];
  float* out = (float*)d_out;

  const long NT = 256L * 180;  // 46080 tokens
  float* ws = (float*)d_ws;
  // arena (floats)
  float* X    = ws;                      // 46080*64
  float* LN   = X    + NT * 64;          // 46080*64
  float* QKV  = LN   + NT * 64;          // 46080*2304
  float* S    = QKV  + NT * 2304;        // 3072*180*180
  float* O    = S    + 3072L * 180 * 180;// 46080*768
  float* FFH  = O    + NT * 768;         // 46080*256
  float* G    = FFH  + NT * 256;         // 256*64*180
  float* PC   = G    + 256L * 64 * 180;  // 256*256*4*8
  float* CAPS = PC   + 256L * 256 * 32;  // 256*1024*8
  float* U    = CAPS + 256L * 1024 * 8;  // 256*1024*48
  float* BB   = U    + 256L * 1024 * 48; // 256*1024*3
  float* Cc   = BB   + 256L * 1024 * 3;  // 256*1024*3
  float* SV   = Cc   + 256L * 1024 * 3;  // 256*3*16
  float* VV   = SV   + 768 * 16;         // 256*3*16

  // conv scratch reuses transformer regions (all dead before transformer)
  float* c1o = QKV;  // 256*64*40*72
  float* p1o = S;    // 256*64*20*36
  float* c2o = O;    // 256*64*20*36
  float* p2o = FFH;  // 256*64*10*18

  // ---- ConvEmbed ----
  {
    long n1 = 256L * 64 * 40 * 72;
    conv3x3_kernel<<<cdiv(n1, 256), 256, 0, stream>>>(x, conv1_w, nullptr, c1o,
        256, 3, 64, 40, 72, 40, 72, 1, 1, 1);
    long np1 = 256L * 64 * 20 * 36;
    maxpool_kernel<<<cdiv(np1, 256), 256, 0, stream>>>(c1o, p1o, 256L * 64, 40, 72, 20, 36);
    conv3x3_kernel<<<cdiv(np1, 256), 256, 0, stream>>>(p1o, conv2_w, nullptr, c2o,
        256, 64, 64, 20, 36, 20, 36, 1, 1, 1);
    long np2 = 256L * 64 * 10 * 18;
    maxpool_kernel<<<cdiv(np2, 256), 256, 0, stream>>>(c2o, p2o, 256L * 64, 20, 36, 10, 18);
    tokens_kernel<<<cdiv(NT * 64, 256), 256, 0, stream>>>(p2o, pos_emb, X);
  }

  // ---- Transformer: 12 shared layers applied twice ----
  for (int it = 0; it < 2; ++it) {
    for (int l = 0; l < 12; ++l) {
      layernorm_kernel<<<NT / 8, 256, 0, stream>>>(X, LN, ln1_g + l * 64, ln1_b + l * 64, NT);
      // qkv: [46080,64] @ [64,2304]
      gemm_kernel<false, 0><<<dim3(720, 36), 128, 0, stream>>>(
          LN, 64, qkv_w + (long)l * 64 * 2304, 2304, QKV, 2304,
          nullptr, nullptr, 0, NT, 2304, 64);
      // scores = Q @ K^T per (b,h)
      attn_scores_kernel<<<dim3(3, 3, 3072), 128, 0, stream>>>(QKV, S);
      softmax_kernel<<<(3072L * 180) / 8, 256, 0, stream>>>(S, 3072L * 180);
      // O = P @ V per (b,h)
      attn_o_kernel<<<dim3(3, 1, 3072), 128, 0, stream>>>(S, QKV, O);
      // out proj + bias + residual -> X
      gemm_kernel<false, 3><<<dim3(720, 1), 128, 0, stream>>>(
          O, 768, out_w + (long)l * 768 * 64, 64, X, 64,
          out_b + l * 64, X, 64, NT, 64, 768);
      layernorm_kernel<<<NT / 8, 256, 0, stream>>>(X, LN, ln2_g + l * 64, ln2_b + l * 64, NT);
      // ff1 + bias + gelu
      gemm_kernel<false, 2><<<dim3(720, 4), 128, 0, stream>>>(
          LN, 64, ff1_w + (long)l * 64 * 256, 256, FFH, 256,
          ff1_b + l * 256, nullptr, 0, NT, 256, 64);
      // ff2 + bias + residual -> X
      gemm_kernel<false, 3><<<dim3(720, 1), 128, 0, stream>>>(
          FFH, 256, ff2_w + (long)l * 256 * 64, 64, X, 64,
          ff2_b + l * 64, X, 64, NT, 64, 256);
    }
  }

  // ---- CapsNet head ----
  g_kernel<<<cdiv(256L * 64 * 180, 256), 256, 0, stream>>>(X, G);
  // PrimaryCaps conv 64->256, k3 s2 p0 -> [256,256,4,8]
  long npc = 256L * 256 * 4 * 8;
  conv3x3_kernel<<<cdiv(npc, 256), 256, 0, stream>>>(G, pcaps_w, pcaps_b, PC,
      256, 64, 256, 10, 18, 4, 8, 2, 0, 0);
  caps_kernel<<<cdiv(256L * 1024, 256), 256, 0, stream>>>(PC, CAPS);
  u_kernel<<<cdiv(256L * 1024 * 48, 256), 256, 0, stream>>>(CAPS, caps_w, U);

  // routing: init bb from b_route; c=softmax(bb); v=squash(sum c*u); 3 iters
  route_init<<<cdiv(256L * 1024 * 3, 256), 256, 0, stream>>>(b_route, BB);
  route_c<<<cdiv(256L * 1024, 256), 256, 0, stream>>>(BB, Cc);
  route_s<<<cdiv(256L * 48, 256), 256, 0, stream>>>(Cc, U, SV);
  route_squash<<<3, 256, 0, stream>>>(SV, VV);
  for (int r = 0; r < 3; ++r) {
    route_bb<<<cdiv(256L * 1024, 256), 256, 0, stream>>>(U, VV, BB);
    route_c<<<cdiv(256L * 1024, 256), 256, 0, stream>>>(BB, Cc);
    route_s<<<cdiv(256L * 48, 256), 256, 0, stream>>>(Cc, U, SV);
    route_squash<<<3, 256, 0, stream>>>(SV, VV);
  }
  vnorm_kernel<<<3, 256, 0, stream>>>(VV, out);
  (void)in_sizes; (void)n_in; (void)out_size; (void)ws_size;
}
